// GraphNet_55430847922682
// MI455X (gfx1250) — matrix operand order repaired and checked by reading, not compiled, not run on hardware
//
#include <hip/hip_runtime.h>
#include <hip/hip_bf16.h>

typedef __attribute__((ext_vector_type(16))) _Float16 v16h;
typedef __attribute__((ext_vector_type(8)))  float    v8f;

#define GV 4
#define NN 204800
#define DD 128
#define EG 3276800
#define BB 128
#define HH 16
#define LL 8
#define CC 2
#define NEG_SLOPE 0.2f

#define EPV (EG + NN)            // edges per view incl. self-loops
#define TILES_PER_VIEW (NN / 16) // 12800
#define TPW 8                    // tiles per wave in the GEMM kernel

// ---------- helpers ----------
__device__ __forceinline__ unsigned enc_f32(float f) {
    unsigned u = __float_as_uint(f);
    return (u & 0x80000000u) ? ~u : (u | 0x80000000u);
}
__device__ __forceinline__ float dec_f32(unsigned u) {
    u = (u & 0x80000000u) ? (u & 0x7FFFFFFFu) : ~u;
    return __uint_as_float(u);
}
__device__ __forceinline__ float leaky(float e) {
    return e > 0.f ? e : NEG_SLOPE * e;
}

// ---------- zero-fill scratch accumulators ----------
__global__ void fill_zero_u32(unsigned* __restrict__ p, long n) {
    long i = (long)blockIdx.x * blockDim.x + threadIdx.x;
    long stride = (long)gridDim.x * blockDim.x;
    for (; i < n; i += stride) p[i] = 0u;
}

// ---------- kernel 1: xp = x @ W_gat (WMMA) + fused a_src/a_dst ----------
__global__ __launch_bounds__(256) void gemm_xp_wmma(
        const float* __restrict__ x,        // [G,N,D]
        const float* __restrict__ Wg,       // [G,D,H]
        const float* __restrict__ att_src,  // [G,H]
        const float* __restrict__ att_dst,  // [G,H]
        float* __restrict__ xp,             // [G,N,H]
        float* __restrict__ as_,            // [G,N]
        float* __restrict__ ad_)            // [G,N]
{
    const int lane = threadIdx.x & 31;
    const int wave = threadIdx.x >> 5;
    const int wid  = blockIdx.x * 8 + wave;          // 8 waves / block
    const int wavesPerView = TILES_PER_VIEW / TPW;   // 1600
    const int g     = wid / wavesPerView;
    const int tile0 = (wid % wavesPerView) * TPW;
    if (g >= GV) return;                             // wave-uniform guard

    const int mn = lane & 15;    // A: row m ; B/D: column n
    const int hi = lane >> 4;    // half-wave selector

    // ---- hoisted B fragments: W_gat[g] as f16, 32x16 layout per K-block ----
    const float* wbase = Wg + (long)g * DD * HH;
    v16h bfrag[4];
#pragma unroll
    for (int kb = 0; kb < 4; ++kb) {
        const float* pb = wbase + (long)(kb * 32 + hi * 16) * HH + mn;
#pragma unroll
        for (int j = 0; j < 16; ++j) bfrag[kb][j] = (_Float16)pb[j * HH];
    }
    const float att_s = att_src[g * HH + mn];
    const float att_d = att_dst[g * HH + mn];

    for (int tt = 0; tt < TPW; ++tt) {
        const int tile = tile0 + tt;
        const long nbase = (long)g * NN + (long)tile * 16;
        const float* xrow = x + (nbase + mn) * DD;

        v8f acc = {};
#pragma unroll
        for (int kb = 0; kb < 4; ++kb) {
            const int k0 = kb * 32;
            // A fragment: contiguous K order (K = k0 + 16*hi + s), matching B
            v16h a;
            const float* pa = xrow + k0 + hi * 16;
#pragma unroll
            for (int j = 0; j < 8; ++j) {
                a[j]     = (_Float16)pa[j];        // K = k0 + 16*hi + j
                a[j + 8] = (_Float16)pa[j + 8];    // K = k0 + 16*hi + j + 8
            }
            acc = __builtin_amdgcn_wmma_f32_16x16x32_f16(
                /*neg_a=*/false, a, /*neg_b=*/false, bfrag[kb],
                /*c_mod=*/(short)0, acc, /*reuse_a=*/false, /*reuse_b=*/false);
        }

        // store D: lane holds column n = mn, rows r + 8*hi
        float* orow = xp + nbase * HH;
#pragma unroll
        for (int r = 0; r < 8; ++r)
            orow[(long)(r + 8 * hi) * HH + mn] = acc[r];

        // fused attention projections: reduce over the 16 columns (lanes)
#pragma unroll
        for (int r = 0; r < 8; ++r) {
            float vs = acc[r] * att_s;
            float vd = acc[r] * att_d;
#pragma unroll
            for (int m = 1; m < 16; m <<= 1) {   // stays inside each 16-lane half
                vs += __shfl_xor(vs, m, 32);
                vd += __shfl_xor(vd, m, 32);
            }
            if (mn == 0) {
                as_[nbase + r + 8 * hi] = vs;
                ad_[nbase + r + 8 * hi] = vd;
            }
        }
    }
}

// ---------- edge index helper ----------
__device__ __forceinline__ void edge_sd(const int* __restrict__ ei, int g, long j,
                                        int& s, int& d) {
    if (j < EG) {
        s = ei[(long)g * 2 * EG + j];
        d = ei[(long)g * 2 * EG + EG + j];
    } else {
        s = d = (int)(j - EG);   // self loop
    }
}

// ---------- kernel 2: segment max over dst ----------
__global__ void edge_max(const int* __restrict__ ei,
                         const float* __restrict__ as_,
                         const float* __restrict__ ad_,
                         unsigned* __restrict__ mmax)
{
    long t = (long)blockIdx.x * blockDim.x + threadIdx.x;
    if (t >= (long)GV * EPV) return;
    const int g = (int)(t / EPV);
    const long j = t % EPV;
    int s, d; edge_sd(ei, g, j, s, d);
    const float e = leaky(as_[(long)g * NN + s] + ad_[(long)g * NN + d]);
    atomicMax(&mmax[(long)g * NN + d], enc_f32(e));
}

// ---------- kernel 3: exp / z-sum / unnormalized feature scatter ----------
// One feature lane per thread: thread t -> edge (t>>4), feature (t&15).
// Each atomic VMEM instruction then touches ~2 consecutive 64B dst rows
// instead of 32 random cachelines; xp[src] gather is a coalesced 64B read.
__global__ void edge_scatter(const int* __restrict__ ei,
                             const float* __restrict__ as_,
                             const float* __restrict__ ad_,
                             const unsigned* __restrict__ mmax,
                             const float* __restrict__ xp,
                             float* __restrict__ z,
                             float* __restrict__ outacc)
{
    long t = (long)blockIdx.x * blockDim.x + threadIdx.x;
    if (t >= (long)GV * EPV * HH) return;
    const int  h    = (int)(t & (HH - 1));
    const long eidx = t >> 4;
    const int  g    = (int)(eidx / EPV);
    const long j    = eidx % EPV;
    int s, d; edge_sd(ei, g, j, s, d);
    const long gi = (long)g * NN;
    const float e  = leaky(as_[gi + s] + ad_[gi + d]);
    const float ex = __expf(e - dec_f32(mmax[gi + d]));
    if (h == 0) atomicAdd(&z[gi + d], ex);
    atomicAdd(&outacc[(gi + d) * HH + h], ex * xp[(gi + s) * HH + h]);
}

// ---------- kernel 4: normalize, bias, score, pooled scatter ----------
__global__ void node_finalize(const float* __restrict__ outacc,
                              const float* __restrict__ z,
                              const float* __restrict__ b_gat,
                              const float* __restrict__ W_post,
                              const float* __restrict__ b_post,
                              const int* __restrict__ ptr,
                              float* __restrict__ pooled)
{
    long i = (long)blockIdx.x * blockDim.x + threadIdx.x;
    if (i >= (long)GV * NN) return;
    const int g = (int)(i / NN);
    const int n = (int)(i % NN);
    const float inv = 1.f / z[i];
    const float* o = outacc + i * HH;
    float score = b_post[g];
#pragma unroll
    for (int h = 0; h < HH; ++h) {
        const float val = o[h] * inv + b_gat[g * HH + h];
        score += val * W_post[g * HH + h];
    }
    // binary search: bidx s.t. ptr[bidx] <= n < ptr[bidx+1]
    int lo = 0, hiB = BB;
    while (lo < hiB) {
        int mid = (lo + hiB) >> 1;
        if (ptr[mid + 1] <= n) lo = mid + 1; else hiB = mid;
    }
    atomicAdd(&pooled[(long)lo * GV + g], score);
}

// ---------- kernel 5: MLP head (one block, one row per thread) ----------
__global__ void mlp_head(const float* __restrict__ pooled,
                         const int* __restrict__ ptr,
                         const float* __restrict__ W0,  const float* __restrict__ b0,
                         const float* __restrict__ Wm,  const float* __restrict__ bm,
                         const float* __restrict__ Wo,  const float* __restrict__ bo,
                         float* __restrict__ out)
{
    const int t = threadIdx.x;
    if (t >= BB) return;
    const float inv_cnt = 1.f / (float)(ptr[t + 1] - ptr[t]);
    float p[GV];
#pragma unroll
    for (int g = 0; g < GV; ++g) p[g] = pooled[(long)t * GV + g] * inv_cnt;

    float h[HH];
#pragma unroll
    for (int jj = 0; jj < HH; ++jj) {
        float acc = b0[jj];
#pragma unroll
        for (int g = 0; g < GV; ++g) acc += p[g] * W0[g * HH + jj];
        h[jj] = acc;
    }
    float h2[HH];
    for (int l = 0; l < LL; ++l) {
        const float* W = Wm + (long)l * HH * HH;
        const float* b = bm + (long)l * HH;
#pragma unroll
        for (int jj = 0; jj < HH; ++jj) {
            float acc = b[jj];
#pragma unroll
            for (int ii = 0; ii < HH; ++ii) acc += h[ii] * W[ii * HH + jj];
            h2[jj] = acc;
        }
#pragma unroll
        for (int jj = 0; jj < HH; ++jj) h[jj] = h2[jj];
    }
#pragma unroll
    for (int c = 0; c < CC; ++c) {
        float acc = bo[c];
#pragma unroll
        for (int ii = 0; ii < HH; ++ii) acc += h[ii] * Wo[ii * CC + c];
        out[t * CC + c] = acc;
    }
}

// ---------- launch ----------
extern "C" void kernel_launch(void* const* d_in, const int* in_sizes, int n_in,
                              void* d_out, int out_size, void* d_ws, size_t ws_size,
                              hipStream_t stream) {
    const float* x        = (const float*)d_in[0];
    const int*   ei       = (const int*)  d_in[1];
    const int*   ptr      = (const int*)  d_in[2];
    const float* W_gat    = (const float*)d_in[3];
    const float* att_src  = (const float*)d_in[4];
    const float* att_dst  = (const float*)d_in[5];
    const float* b_gat    = (const float*)d_in[6];
    const float* W_post   = (const float*)d_in[7];
    const float* b_post   = (const float*)d_in[8];
    const float* W0       = (const float*)d_in[9];
    const float* b0       = (const float*)d_in[10];
    const float* W_mid    = (const float*)d_in[11];
    const float* b_mid    = (const float*)d_in[12];
    const float* W_out    = (const float*)d_in[13];
    const float* b_out    = (const float*)d_in[14];
    float* out = (float*)d_out;

    // workspace carve-up (all 4-byte words)
    float*    xp     = (float*)d_ws;                                   // G*N*H
    float*    as_    = xp   + (size_t)GV * NN * HH;                    // G*N
    float*    ad_    = as_  + (size_t)GV * NN;                         // G*N
    unsigned* mmax   = (unsigned*)(ad_ + (size_t)GV * NN);             // G*N
    float*    z      = (float*)(mmax + (size_t)GV * NN);               // G*N
    float*    outacc = z    + (size_t)GV * NN;                         // G*N*H
    float*    pooled = outacc + (size_t)GV * NN * HH;                  // B*G

    // zero accumulators (mmax, z, outacc, pooled)
    const long zero_words = (long)GV * NN * (2 + HH) + (long)BB * GV;
    fill_zero_u32<<<4096, 256, 0, stream>>>(mmax, zero_words);

    // 1) GEMM xp = x @ W_gat (WMMA) + fused a_src/a_dst
    const int gemm_blocks = (GV * TILES_PER_VIEW) / (8 * TPW);         // 800
    gemm_xp_wmma<<<gemm_blocks, 256, 0, stream>>>(x, W_gat, att_src, att_dst,
                                                  xp, as_, ad_);

    // 2) per-dst softmax max
    const long TE = (long)GV * EPV;
    edge_max<<<(int)((TE + 255) / 256), 256, 0, stream>>>(ei, as_, ad_, mmax);

    // 3) exp + z + unnormalized feature scatter (one feature lane per thread)
    const long TS = TE * HH;
    edge_scatter<<<(int)((TS + 255) / 256), 256, 0, stream>>>(ei, as_, ad_, mmax, xp, z, outacc);

    // 4) normalize + score + ragged mean-pool (sum part)
    node_finalize<<<(GV * NN + 255) / 256, 256, 0, stream>>>(outacc, z, b_gat, W_post,
                                                             b_post, ptr, pooled);

    // 5) MLP head -> [128, 2]
    mlp_head<<<1, 128, 0, stream>>>(pooled, ptr, W0, b0, W_mid, b_mid, W_out, b_out, out);
}
